// SpatialTransformerNetwork_20272245637164
// MI455X (gfx1250) — matrix operand order repaired and checked by reading, not compiled, not run on hardware
//
#include <hip/hip_runtime.h>

typedef __attribute__((ext_vector_type(16))) _Float16 v16h;
typedef __attribute__((ext_vector_type(8)))  _Float16 v8h;
typedef __attribute__((ext_vector_type(4)))  _Float16 v4h;
typedef __attribute__((ext_vector_type(2)))  _Float16 v2h;
typedef __attribute__((ext_vector_type(8)))  float    v8f;

#define SH16(lo, hi) __builtin_shufflevector(lo, hi, 0,1,2,3,4,5,6,7,8,9,10,11,12,13,14,15)
#define SH8(lo, hi)  __builtin_shufflevector(lo, hi, 0,1,2,3,4,5,6,7)

__device__ __forceinline__ v8f wmma_f32_16x16x32_f16(v16h a, v16h b, v8f c) {
  return __builtin_amdgcn_wmma_f32_16x16x32_f16(false, a, false, b, (short)0, c, false, false);
}

// ---------------------------------------------------------------------------
// Prep: transpose + f16-convert all weights into WMMA-B-friendly [n][k] tiles.
//   wT1 : [16][224]   conv1, kpad = dy*32 + dx*4 + c (dx<7, c<3 valid, else 0)
//   wT2 : [32][416]   conv2, k = (dy*5+dx)*16 + c, zero for k>=400
//   wTfc: [64][25088] fc1,  zero for n>=50
// ---------------------------------------------------------------------------
__global__ __launch_bounds__(256)
void stn_prep_weights(const float* __restrict__ w1, const float* __restrict__ w2,
                      const float* __restrict__ wfc,
                      _Float16* __restrict__ wT1, _Float16* __restrict__ wT2,
                      _Float16* __restrict__ wTfc)
{
  const int nT1 = 16 * 224, nT2 = 32 * 416, nTf = 64 * 25088;
  int i = blockIdx.x * 256 + threadIdx.x;
  if (i < nT1) {
    int n = i / 224, kp = i % 224;
    int dy = kp >> 5, j = kp & 31, dx = j >> 2, c = j & 3;
    float v = (dx < 7 && c < 3) ? w1[((dy * 7 + dx) * 3 + c) * 16 + n] : 0.f;
    wT1[i] = (_Float16)v;
  } else if (i < nT1 + nT2) {
    int idx = i - nT1;
    int n = idx / 416, k = idx % 416;
    float v = (k < 400) ? w2[k * 32 + n] : 0.f;
    wT2[idx] = (_Float16)v;
  } else if (i < nT1 + nT2 + nTf) {
    int idx = i - nT1 - nT2;
    int n = idx / 25088, k = idx % 25088;
    float v = (n < 50) ? wfc[(size_t)k * 50 + n] : 0.f;
    wTfc[idx] = (_Float16)v;
  }
}

// ---------------------------------------------------------------------------
// Kernel 1: conv1 (7x7x3->16) + ReLU + 2x2 maxpool, fused.
// Grid (61,128); 8 waves; wave = 16-pos x 16-ch tile, two conv rows, pooled.
// LDS input layout: [8 rows][129 cols][4 ch] f16 (c=3 and col 128 zero-padded)
// K padded to 224 (= 7 rows * 32): one WMMA K-step per conv tap row.
// A fragment = 4x ds_load_b64 (8B aligned); B fragment = 2x ds_load_b128.
// ---------------------------------------------------------------------------
__global__ __launch_bounds__(256)
void stn_conv1_pool(const float* __restrict__ x, const _Float16* __restrict__ wT1g,
                    const float* __restrict__ bias, _Float16* __restrict__ out1)
{
  __shared__ alignas(16) _Float16 s_in[8 * 516];   // 516 = 129*4 halfs/row
  __shared__ alignas(16) _Float16 s_w[16 * 224];   // [n][kpad]
  const int py = blockIdx.x, b = blockIdx.y, tid = threadIdx.x;

  // Stage 8 input rows (2py..2py+7), converting layout [128][3]f32 -> [129][4]f16
  const float* src = x + (size_t)(b * 128 + 2 * py) * 384;
  for (int i2 = tid; i2 < 2064; i2 += 256) {
    int h0 = i2 * 2;
    int row = h0 / 516, rem = h0 % 516;
    int col = rem >> 2, cp = rem & 3;              // cp in {0,2}
    v2h v;
    v.x = (col < 128 && cp < 3)     ? (_Float16)src[row * 384 + col * 3 + cp]     : (_Float16)0.f;
    v.y = (col < 128 && cp + 1 < 3) ? (_Float16)src[row * 384 + col * 3 + cp + 1] : (_Float16)0.f;
    *(v2h*)(s_in + h0) = v;
  }
  for (int i = tid; i < 448; i += 256)             // 3584 halfs via b128
    ((v8h*)s_w)[i] = ((const v8h*)wT1g)[i];
  __syncthreads();

  const int wv = tid >> 5, lane = tid & 31;
  const int mrow = lane & 15, hi = lane >> 4;
  const int x0 = wv * 16;
  int xm = x0 + mrow; if (xm > 121) xm = 121;      // clamped lanes masked at store

  // B fragments: elem e(0..7) -> K = kb*32 + hi*8 + e ; e(8..15) -> +16
  v16h Bf[7];
  {
    const int n = lane & 15;
#pragma unroll
    for (int kb = 0; kb < 7; ++kb) {
      const v8h* p = (const v8h*)(s_w + n * 224 + kb * 32 + hi * 8);
      Bf[kb] = SH16(p[0], p[2]);
    }
  }

  v8f acc[2];
#pragma unroll
  for (int yr = 0; yr < 2; ++yr) {
    v8f c = {};
#pragma unroll
    for (int kb = 0; kb < 7; ++kb) {
      // A elem e(0..7) -> K = kb*32 + hi*8 + e ; e(8..15) -> +16
      const _Float16* ap = s_in + (yr + kb) * 516 + xm * 4 + hi * 8;
      const v4h* q1 = (const v4h*)ap;
      const v4h* q2 = (const v4h*)(ap + 16);
      v8h lo = SH8(q1[0], q1[1]);
      v8h hb = SH8(q2[0], q2[1]);
      c = wmma_f32_16x16x32_f16(SH16(lo, hb), Bf[kb], c);
    }
    acc[yr] = c;
  }

  // C layout: lane -> channel; vgpr r -> position r + 8*hi. Pool in registers.
  const int ch = lane & 15;
  const float bv = bias[ch];
  const int pxb = (x0 >> 1) + (hi ? 4 : 0);
  _Float16* orow = out1 + (size_t)(b * 61 + py) * (61 * 16);
#pragma unroll
  for (int i = 0; i < 4; ++i) {
    int px = pxb + i;
    if (px < 61) {
      float a0 = fmaxf(acc[0][2 * i]     + bv, 0.f);
      float a1 = fmaxf(acc[0][2 * i + 1] + bv, 0.f);
      float b0 = fmaxf(acc[1][2 * i]     + bv, 0.f);
      float b1 = fmaxf(acc[1][2 * i + 1] + bv, 0.f);
      orow[px * 16 + ch] = (_Float16)fmaxf(fmaxf(a0, a1), fmaxf(b0, b1));
    }
  }
}

// ---------------------------------------------------------------------------
// Kernel 2: conv2 (5x5x16->32) + ReLU + 2x2 maxpool.
// Grid (28,128); wave -> (pos tile 0..3, chan half 0..1). C=16 makes every
// A chunk 16B-aligned: 2x ds_load_b128 per fragment, all offsets compile-time.
// ---------------------------------------------------------------------------
__global__ __launch_bounds__(256)
void stn_conv2_pool(const _Float16* __restrict__ in1, const _Float16* __restrict__ wT2g,
                    const float* __restrict__ bias, _Float16* __restrict__ out2)
{
  __shared__ alignas(16) _Float16 s_in[6 * 976];   // 6 input rows [61][16]
  __shared__ alignas(16) _Float16 s_w[32 * 416];   // [n][k]
  const int py = blockIdx.x, b = blockIdx.y, tid = threadIdx.x;

  const _Float16* src = in1 + (size_t)(b * 61 + 2 * py) * 976;
  for (int i = tid; i < 732; i += 256)             // 5856 halfs via b128
    ((v8h*)s_in)[i] = ((const v8h*)src)[i];
  for (int i = tid; i < 1664; i += 256)            // 13312 halfs via b128
    ((v8h*)s_w)[i] = ((const v8h*)wT2g)[i];
  __syncthreads();

  const int wv = tid >> 5, lane = tid & 31;
  const int t = wv & 3, g = wv >> 2;
  const int mrow = lane & 15, hi = lane >> 4;
  const int x0 = t * 16;
  int xm = x0 + mrow; if (xm > 56) xm = 56;
  const int n2 = g * 16 + (lane & 15);

  const v8h zv = {};
  v8f c0 = {}, c1 = {};
#pragma unroll
  for (int kb = 0; kb < 13; ++kb) {
    const v8h* pb = (const v8h*)(s_w + n2 * 416 + kb * 32 + hi * 8);
    const v16h bf = SH16(pb[0], pb[2]);
    const int t1 = 2 * kb, t2 = 2 * kb + 1;        // taps (compile-time)
    const int dy1 = t1 / 5, dx1 = t1 % 5;
    const int dy2 = t2 / 5, dx2 = t2 % 5;
#pragma unroll
    for (int yr = 0; yr < 2; ++yr) {
      const v8h* q1 = (const v8h*)(s_in + (yr + dy1) * 976 + (xm + dx1) * 16 + hi * 8);
      v8h lo = q1[0];
      v8h hb;
      if (t2 < 25) {
        const v8h* q2 = (const v8h*)(s_in + (yr + dy2) * 976 + (xm + dx2) * 16 + hi * 8);
        hb = q2[0];
      } else {
        hb = zv;                                   // K pad 400..415 (B is zero too)
      }
      v16h a = SH16(lo, hb);
      if (yr == 0) c0 = wmma_f32_16x16x32_f16(a, bf, c0);
      else         c1 = wmma_f32_16x16x32_f16(a, bf, c1);
    }
  }

  const float bv = bias[n2];
  const int pxb = (x0 >> 1) + (hi ? 4 : 0);
  _Float16* orow = out2 + (size_t)(b * 28 + py) * (28 * 32);
#pragma unroll
  for (int i = 0; i < 4; ++i) {
    int px = pxb + i;
    if (px < 28) {
      float p0 = fmaxf(c0[2 * i]     + bv, 0.f);
      float p1 = fmaxf(c0[2 * i + 1] + bv, 0.f);
      float q0 = fmaxf(c1[2 * i]     + bv, 0.f);
      float q1 = fmaxf(c1[2 * i + 1] + bv, 0.f);
      orow[px * 32 + n2] = (_Float16)fmaxf(fmaxf(p0, p1), fmaxf(q0, q1));
    }
  }
}

// ---------------------------------------------------------------------------
// Kernel 3: fc1 GEMM [128 x 25088]f16 @ [25088 x 64]f16 (+bias, ReLU).
// 4 blocks x 8 waves = 32 wave-tiles; B pre-transposed => 2x b128 global loads.
// ---------------------------------------------------------------------------
__global__ __launch_bounds__(256)
void stn_fc1(const _Float16* __restrict__ a_in, const _Float16* __restrict__ wTfc,
             const float* __restrict__ b1, float* __restrict__ h_out)
{
  const int tid = threadIdx.x;
  const int gw = blockIdx.x * 8 + (tid >> 5);
  const int lane = tid & 31;
  const int mtile = gw >> 2, ntile = gw & 3;
  const int mrow = lane & 15, hi = lane >> 4;
  const int n = ntile * 16 + (lane & 15);
  const _Float16* arow = a_in + (size_t)(mtile * 16 + mrow) * 25088;
  const _Float16* brow = wTfc + (size_t)n * 25088;

  v8f acc = {};
#pragma unroll 8
  for (int kb = 0; kb < 784; ++kb) {
    const v8h* pa = (const v8h*)(arow + kb * 32 + hi * 8);
    const v8h* pb = (const v8h*)(brow + kb * 32 + hi * 8);
    v16h a  = SH16(pa[0], pa[2]);
    v16h bf = SH16(pb[0], pb[2]);
    acc = wmma_f32_16x16x32_f16(a, bf, acc);
  }

  const bool nok = (n < 50);
  const float bv = nok ? b1[n] : 0.f;
#pragma unroll
  for (int r = 0; r < 8; ++r) {
    int m = mtile * 16 + hi * 8 + r;
    h_out[m * 64 + n] = nok ? fmaxf(acc[r] + bv, 0.f) : 0.f;
  }
}

// ---------------------------------------------------------------------------
// Kernel 4: fc2 -> theta[128][6]
// ---------------------------------------------------------------------------
__global__ __launch_bounds__(256)
void stn_fc2(const float* __restrict__ h, const float* __restrict__ w2,
             const float* __restrict__ b2, float* __restrict__ theta)
{
  for (int idx = threadIdx.x; idx < 768; idx += 256) {
    int m = idx / 6, j = idx % 6;
    float s = b2[j];
#pragma unroll 10
    for (int nn = 0; nn < 50; ++nn) s += h[m * 64 + nn] * w2[nn * 6 + j];
    theta[m * 6 + j] = s;
  }
}

// ---------------------------------------------------------------------------
// Kernel 5: affine grid + bilinear sample. Bandwidth stage; input L2-resident.
// ---------------------------------------------------------------------------
__global__ __launch_bounds__(256)
void stn_sample(const float* __restrict__ x, const float* __restrict__ theta,
                float* __restrict__ out)
{
  const int gid = blockIdx.x * 256 + threadIdx.x;
  const int b   = gid >> 14;
  const int rem = gid & 16383;
  const int oy  = rem >> 7, ox = rem & 127;

  __shared__ float th[6];
  if (threadIdx.x < 6) th[threadIdx.x] = theta[b * 6 + threadIdx.x];
  __syncthreads();

  const float gx = ox * (2.0f / 127.0f) - 1.0f;
  const float gy = oy * (2.0f / 127.0f) - 1.0f;
  const float sx = th[0] * gx + th[1] * gy + th[2];
  const float sy = th[3] * gx + th[4] * gy + th[5];
  const float xp = (sx + 1.0f) * 63.5f;
  const float yp = (sy + 1.0f) * 63.5f;
  const float x0f = floorf(xp), y0f = floorf(yp);
  const float wx = xp - x0f, wy = yp - y0f;
  int x0i = (int)x0f; x0i = x0i < 0 ? 0 : (x0i > 127 ? 127 : x0i);
  int x1i = x0i + 1;  x1i = x1i > 127 ? 127 : x1i;
  int y0i = (int)y0f; y0i = y0i < 0 ? 0 : (y0i > 127 ? 127 : y0i);
  int y1i = y0i + 1;  y1i = y1i > 127 ? 127 : y1i;

  const float* img = x + (size_t)b * 49152;
  const float* Ia = img + (y0i * 128 + x0i) * 3;
  const float* Ib = img + (y1i * 128 + x0i) * 3;
  const float* Ic = img + (y0i * 128 + x1i) * 3;
  const float* Id = img + (y1i * 128 + x1i) * 3;
  const float w00 = (1.f - wx) * (1.f - wy);
  const float w01 = (1.f - wx) * wy;
  const float w10 = wx * (1.f - wy);
  const float w11 = wx * wy;
  float* o = out + (size_t)gid * 3;
#pragma unroll
  for (int c = 0; c < 3; ++c)
    o[c] = Ia[c] * w00 + Ib[c] * w01 + Ic[c] * w10 + Id[c] * w11;
}

// ---------------------------------------------------------------------------
extern "C" void kernel_launch(void* const* d_in, const int* in_sizes, int n_in,
                              void* d_out, int out_size, void* d_ws, size_t ws_size,
                              hipStream_t stream)
{
  (void)in_sizes; (void)n_in; (void)out_size; (void)ws_size;
  const float* x    = (const float*)d_in[0];
  const float* wc1  = (const float*)d_in[1];
  const float* bc1  = (const float*)d_in[2];
  const float* wc2  = (const float*)d_in[3];
  const float* bc2  = (const float*)d_in[4];
  const float* wfc1 = (const float*)d_in[5];
  const float* bfc1 = (const float*)d_in[6];
  const float* wfc2 = (const float*)d_in[7];
  const float* bfc2 = (const float*)d_in[8];

  char* ws = (char*)d_ws;
  _Float16* o1   = (_Float16*)(ws);                  // 128*61*61*16 f16 = 15,241,216 B
  _Float16* o2   = (_Float16*)(ws + 15236096);       // 128*28*28*32 f16 =  6,422,528 B
  float*    hbuf = (float*)(ws + 21658624);          // 128*64 f32       =     32,768 B
  float*    thet = (float*)(ws + 21691392);          // 128*6  f32       =      3,072 B
  _Float16* wT1  = (_Float16*)(ws + 21694464);       // 16*224  f16      =      7,168 B
  _Float16* wT2  = (_Float16*)(ws + 21701632);       // 32*416  f16      =     26,624 B
  _Float16* wTfc = (_Float16*)(ws + 21728256);       // 64*25088 f16     =  3,211,264 B

  const int prep_total = 16 * 224 + 32 * 416 + 64 * 25088;   // 1,622,528
  stn_prep_weights<<<(prep_total + 255) / 256, 256, 0, stream>>>(wc1, wc2, wfc1, wT1, wT2, wTfc);
  stn_conv1_pool<<<dim3(61, 128), 256, 0, stream>>>(x, wT1, bc1, o1);
  stn_conv2_pool<<<dim3(28, 128), 256, 0, stream>>>(o1, wT2, bc2, o2);
  stn_fc1<<<4, 256, 0, stream>>>(o2, wTfc, bfc1, hbuf);
  stn_fc2<<<1, 256, 0, stream>>>(hbuf, wfc2, bfc2, thet);
  stn_sample<<<8192, 256, 0, stream>>>(x, thet, (float*)d_out);
}
